// HiResPrecipNet_temporal_89223650607425
// MI455X (gfx1250) — matrix operand order repaired and checked by reading, not compiled, not run on hardware
//
#include <hip/hip_runtime.h>
#include <hip/hip_bf16.h>

// ---------------- problem constants ----------------
#define N_LOW   10000
#define N_HIGH  100000
#define LOW_IN  25
#define PERIODS 25
#define E_LL    80000
#define E_LH    (N_HIGH*8)
#define E_HH    800000
#define EPSBN   1e-5f

#define DIVUP(a,b) (((a)+(b)-1)/(b))

// ---------------- WMMA types (gfx1250, wave32) ----------------
typedef __bf16 bf16x16 __attribute__((ext_vector_type(16)));
typedef float  f32x8   __attribute__((ext_vector_type(8)));

union BFrag { bf16x16 v; unsigned short u[16]; uint4 q[2]; };
union CFrag { f32x8   v; float          e[8];  };

// f32 -> bf16 round-to-nearest-even (bit pattern)
__device__ inline unsigned short f2bf(float f) {
  unsigned int u = __float_as_uint(f);
  u += 0x7fffu + ((u >> 16) & 1u);
  return (unsigned short)(u >> 16);
}

__device__ inline void atomicMaxFloat(float* addr, float val) {
  if (val >= 0.0f) atomicMax((int*)addr, __float_as_int(val));
  else             atomicMin((unsigned int*)addr, __float_as_uint(val));
}

// A fragment: 16x32 bf16, row-major source with ld=32 (64B rows, 16B aligned groups).
// ISA 7.12.2: lane l<16 holds row M=l, K={0..7,16..23}; lane l+16 holds K={8..15,24..31}.
// Each half is one contiguous 16-byte run -> two global_load_b128 per lane.
__device__ inline void load_afrag(BFrag& a, const unsigned short* __restrict__ A,
                                  int mt, int lane) {
  const uint4* p = (const uint4*)(A + (size_t)(mt * 16 + (lane & 15)) * 32 + ((lane >> 4) << 3));
  a.q[0] = p[0];   // K = kh .. kh+7
  a.q[1] = p[2];   // K = 16+kh .. 16+kh+7
}

// B fragment from pre-packed layout: Bp[nt][lane][16] contiguous bf16 per lane
// (lane l<16: col=nt*16+l, K={0..7,16..23} ; lane l+16: same col, K={8..15,24..31})
__device__ inline void load_bfrag(BFrag& b, const unsigned short* __restrict__ Bp,
                                  int nt, int lane) {
  const uint4* p = (const uint4*)(Bp + (size_t)(((nt << 5) + lane) << 4));
  b.q[0] = p[0];
  b.q[1] = p[1];
}

// pack f32 weight [src_rows, src_cols] (zero-padded to [32, ncols]) into fragment order
__global__ void pack_bfrag(unsigned short* __restrict__ dst, const float* __restrict__ src,
                           int src_rows, int src_cols, int ncols) {
  int t = blockIdx.x * blockDim.x + threadIdx.x;
  if (t >= 32 * ncols) return;          // (ncols/16) tiles * 32 lanes * 16 elems
  int nt   = t >> 9;                    // / 512
  int rem  = t & 511;
  int lane = rem >> 4;
  int j    = rem & 15;
  int k    = ((j >> 3) << 4) + ((lane >> 4) << 3) + (j & 7);    // K index 0..31
  int col  = (nt << 4) + (lane & 15);
  unsigned short v = 0;
  if (k < src_rows && col < src_cols) v = f2bf(src[(size_t)k * src_cols + col]);
  dst[t] = v;
}

// ---------------- generic WMMA GEMM: Y[M,ncols] = A[M,32]bf16 @ B[32,ncols]bf16 + bias ----------------
__global__ __launch_bounds__(256)
void gemm_bf16(const unsigned short* __restrict__ A, const unsigned short* __restrict__ Bp,
               const float* __restrict__ bias, int bias_n,
               float* __restrict__ Y, int ldy, int Mtiles, int ncols, int relu) {
  int lane = threadIdx.x & 31;
  int wid  = (blockIdx.x * blockDim.x + threadIdx.x) >> 5;
  int Ntiles = ncols >> 4;
  if (wid >= Mtiles * Ntiles) return;  // wave-uniform exit: EXEC all-1 for WMMA
  int mt = wid / Ntiles, nt = wid - mt * Ntiles;
  BFrag a, b; CFrag c;
  load_afrag(a, A, mt, lane);
  load_bfrag(b, Bp, nt, lane);
#pragma unroll
  for (int j = 0; j < 8; ++j) c.e[j] = 0.0f;
  c.v = __builtin_amdgcn_wmma_f32_16x16x32_bf16(false, a.v, false, b.v, (short)0, c.v, false, false);
  int col = (nt << 4) + (lane & 15);
  float bv = (col < bias_n) ? bias[col] : 0.0f;
  int rbase = mt * 16 + ((lane >> 4) << 3);
#pragma unroll
  for (int j = 0; j < 8; ++j) {
    float v = c.e[j] + bv;
    if (relu) v = fmaxf(v, 0.0f);
    Y[(size_t)(rbase + j) * ldy + col] = v;
  }
}

// ---------------- misc elementwise kernels ----------------
__global__ void fill_f32(float* __restrict__ p, float v, int n) {
  int t = blockIdx.x * blockDim.x + threadIdx.x;
  if (t < n) p[t] = v;
}

// generic (padding) f32 -> bf16 row-major
__global__ void f32_to_bf16_pad(unsigned short* __restrict__ dst, const float* __restrict__ src,
                                int dst_rows, int dst_cols, int src_rows, int src_cols) {
  int t = blockIdx.x * blockDim.x + threadIdx.x;
  if (t >= dst_rows * dst_cols) return;
  int r = t / dst_cols, c = t - r * dst_cols;
  unsigned short v = 0;
  if (r < src_rows && c < src_cols) v = f2bf(src[(size_t)r * src_cols + c]);
  dst[t] = v;
}

// fast path for big equal-shape arrays: float4 -> ushort4 (b128 load, b64 store)
__global__ void f32_to_bf16_vec4(unsigned short* __restrict__ dst, const float* __restrict__ src,
                                 int n4) {
  int t = blockIdx.x * blockDim.x + threadIdx.x;
  if (t >= n4) return;
  float4 f = ((const float4*)src)[t];
  ushort4 o;
  o.x = f2bf(f.x); o.y = f2bf(f.y); o.z = f2bf(f.z); o.w = f2bf(f.w);
  ((ushort4*)dst)[t] = o;
}

// ---------------- Stage A: A3TGCN on low graph ----------------
__global__ void deg_count(const int* __restrict__ ell, float* __restrict__ deg) {
  int t = blockIdx.x * blockDim.x + threadIdx.x;
  if (t >= E_LL + N_LOW) return;
  int d = (t < E_LL) ? ell[E_LL + t] : (t - E_LL);
  atomicAdd(&deg[d], 1.0f);
}
__global__ void deg_to_dinv(float* __restrict__ deg) {
  int t = blockIdx.x * blockDim.x + threadIdx.x;
  if (t >= N_LOW) return;
  float v = deg[t];
  deg[t] = (v > 0.0f) ? rsqrtf(v) : 0.0f;
}
__global__ void softmax_periods(const float* __restrict__ att, float* __restrict__ probs) {
  if (threadIdx.x != 0 || blockIdx.x != 0) return;
  float m = -1e30f;
  for (int i = 0; i < PERIODS; ++i) m = fmaxf(m, att[i]);
  float e[PERIODS], s = 0.0f;
  for (int i = 0; i < PERIODS; ++i) { e[i] = __expf(att[i] - m); s += e[i]; }
  float inv = 1.0f / s;
  for (int i = 0; i < PERIODS; ++i) probs[i] = e[i] * inv;
}
// xmsg[(n*P+p)*32+c] = dinv[n]^2 * x_low[n,c,p]  (self-loop term), pad cols 25..31 = 0
__global__ void xmsg_init(const float* __restrict__ x_low, const float* __restrict__ dinv,
                          float* __restrict__ xmsg) {
  int t = blockIdx.x * blockDim.x + threadIdx.x;
  if (t >= N_LOW * PERIODS * 32) return;
  int c = t & 31, row = t >> 5;
  int n = row / PERIODS, p = row - n * PERIODS;
  float v = 0.0f;
  if (c < LOW_IN) { float di = dinv[n]; v = di * di * x_low[(size_t)n * 625 + c * 25 + p]; }
  xmsg[t] = v;
}
__global__ void xmsg_scatter(const float* __restrict__ x_low, const int* __restrict__ ell,
                             const float* __restrict__ dinv, float* __restrict__ xmsg) {
  int t = blockIdx.x * blockDim.x + threadIdx.x;
  if (t >= E_LL * PERIODS) return;
  int e = t % E_LL, p = t / E_LL;
  int s = ell[e], d = ell[E_LL + e];
  float w = dinv[s] * dinv[d];
  float* dst = xmsg + (size_t)(d * PERIODS + p) * 32;
  const float* xs = x_low + (size_t)s * 625 + p;
#pragma unroll 5
  for (int c = 0; c < LOW_IN; ++c) atomicAdd(&dst[c], w * xs[c * 25]);
}
// Fused gate: Z=sigmoid(hz@Wlz+blz), Ht=tanh(hh@Wlh+blh), enc[n] += probs[p]*(1-Z)*Ht
__global__ __launch_bounds__(256)
void tgcn_gate_accum(const unsigned short* __restrict__ hzb, const unsigned short* __restrict__ hhb,
                     const unsigned short* __restrict__ wlz, const unsigned short* __restrict__ wlh,
                     const float* __restrict__ blz, const float* __restrict__ blh,
                     const float* __restrict__ probs, float* __restrict__ enc, int Mtiles) {
  int lane = threadIdx.x & 31;
  int wid  = (blockIdx.x * blockDim.x + threadIdx.x) >> 5;
  if (wid >= Mtiles) return;
  BFrag az, ah, b;
  load_afrag(az, hzb, wid, lane);
  load_afrag(ah, hhb, wid, lane);
  for (int nt = 0; nt < 2; ++nt) {
    CFrag cz, ch;
#pragma unroll
    for (int j = 0; j < 8; ++j) { cz.e[j] = 0.0f; ch.e[j] = 0.0f; }
    load_bfrag(b, wlz, nt, lane);
    cz.v = __builtin_amdgcn_wmma_f32_16x16x32_bf16(false, az.v, false, b.v, (short)0, cz.v, false, false);
    load_bfrag(b, wlh, nt, lane);
    ch.v = __builtin_amdgcn_wmma_f32_16x16x32_bf16(false, ah.v, false, b.v, (short)0, ch.v, false, false);
    int col = (nt << 4) + (lane & 15);
    if (col < LOW_IN) {
      float bzv = blz[col], bhv = blh[col];
      int rbase = wid * 16 + ((lane >> 4) << 3);
#pragma unroll
      for (int j = 0; j < 8; ++j) {
        int row = rbase + j;
        int n = row / PERIODS, p = row - n * PERIODS;
        float z  = 1.0f / (1.0f + __expf(-(cz.e[j] + bzv)));
        float ht = tanhf(ch.e[j] + bhv);
        atomicAdd(&enc[n * 32 + col], probs[p] * (1.0f - z) * ht);
      }
    }
  }
}

// ---------------- Stage B: GATv2 downscaler (8 in-edges per high node, 4 heads) ----------------
__global__ void h0_init(const float* __restrict__ z_std, float* __restrict__ h) {
  int t = blockIdx.x * blockDim.x + threadIdx.x;
  if (t >= N_HIGH * 32) return;
  int c = t & 31, n = t >> 5;
  h[t] = (c == 0) ? z_std[n] : 0.0f;
}
__global__ void downscale_gat(const float* __restrict__ xl_low, const int* __restrict__ src_lh,
                              const float* __restrict__ x_high, const float* __restrict__ Wr,
                              const float* __restrict__ br, const float* __restrict__ att,
                              const float* __restrict__ bias, float* __restrict__ h) {
  int n = blockIdx.x * blockDim.x + threadIdx.x;
  if (n >= N_HIGH) return;
  float xh = x_high[n];
  float out[16];
#pragma unroll
  for (int c = 0; c < 16; ++c) out[c] = 0.0f;
  int e0 = n * 8;
  for (int hd = 0; hd < 4; ++hd) {
    float xr[16], atv[16];
#pragma unroll
    for (int q = 0; q < 4; ++q) {
      float4 w = ((const float4*)(Wr + hd * 16))[q];
      float4 bq = ((const float4*)(br + hd * 16))[q];
      float4 aq = ((const float4*)(att + hd * 16))[q];
      xr[4*q+0] = xh * w.x + bq.x; xr[4*q+1] = xh * w.y + bq.y;
      xr[4*q+2] = xh * w.z + bq.z; xr[4*q+3] = xh * w.w + bq.w;
      atv[4*q+0] = aq.x; atv[4*q+1] = aq.y; atv[4*q+2] = aq.z; atv[4*q+3] = aq.w;
    }
    float lg[8], mx = -1e30f;
    for (int k = 0; k < 8; ++k) {
      int s = src_lh[e0 + k];
      const float4* xj = (const float4*)(xl_low + (size_t)s * 64 + hd * 16);
      float acc = 0.0f;
#pragma unroll
      for (int q = 0; q < 4; ++q) {
        float4 xv = xj[q];
        float v0 = xv.x + xr[4*q+0], v1 = xv.y + xr[4*q+1];
        float v2 = xv.z + xr[4*q+2], v3 = xv.w + xr[4*q+3];
        v0 = (v0 > 0.0f) ? v0 : 0.2f * v0; v1 = (v1 > 0.0f) ? v1 : 0.2f * v1;
        v2 = (v2 > 0.0f) ? v2 : 0.2f * v2; v3 = (v3 > 0.0f) ? v3 : 0.2f * v3;
        acc += atv[4*q+0]*v0 + atv[4*q+1]*v1 + atv[4*q+2]*v2 + atv[4*q+3]*v3;
      }
      lg[k] = acc; mx = fmaxf(mx, acc);
    }
    float ssum = 0.0f;
    for (int k = 0; k < 8; ++k) { lg[k] = __expf(lg[k] - mx); ssum += lg[k]; }
    float inv = 1.0f / ssum;
    for (int k = 0; k < 8; ++k) {
      int s = src_lh[e0 + k];
      const float4* xj = (const float4*)(xl_low + (size_t)s * 64 + hd * 16);
      float w = lg[k] * inv;
#pragma unroll
      for (int q = 0; q < 4; ++q) {
        float4 xv = xj[q];
        out[4*q+0] += w * xv.x; out[4*q+1] += w * xv.y;
        out[4*q+2] += w * xv.z; out[4*q+3] += w * xv.w;
      }
    }
  }
  for (int c = 0; c < 16; ++c) h[n * 32 + 1 + c] = out[c] * (1.0f / 32.0f) + bias[c];
}

// ---------------- BatchNorm ----------------
__global__ void bn_reduce(const float* __restrict__ x, int N, int C, int ld,
                          float* __restrict__ sum, float* __restrict__ sumsq) {
  int tid = blockIdx.x * blockDim.x + threadIdx.x;
  int nthreads = gridDim.x * blockDim.x;
  int c = tid % ld;
  if (c >= C) return;
  int stride = nthreads / ld;
  float s = 0.0f, q = 0.0f;
  for (int n = tid / ld; n < N; n += stride) {
    float v = x[(size_t)n * ld + c];
    s += v; q += v * v;
  }
  atomicAdd(&sum[c], s);
  atomicAdd(&sumsq[c], q);
}
__global__ void bn_norm(float* __restrict__ x, int N, int C, int ld,
                        const float* __restrict__ sum, const float* __restrict__ sumsq,
                        const float* __restrict__ g, const float* __restrict__ bb, int relu) {
  int t = blockIdx.x * blockDim.x + threadIdx.x;
  if (t >= N * C) return;
  int c = t % C, n = t / C;
  float mu  = sum[c] / (float)N;
  float var = sumsq[c] / (float)N - mu * mu;
  float v = (x[(size_t)n * ld + c] - mu) * rsqrtf(var + EPSBN) * g[c] + bb[c];
  if (relu) v = fmaxf(v, 0.0f);
  x[(size_t)n * ld + c] = v;
}

// ---------------- high-graph GATv2 edge phases (E_HH edges + N_HIGH self loops) ----------------
__global__ void cnt_count(const int* __restrict__ dst, float* __restrict__ cnt) {
  int t = blockIdx.x * blockDim.x + threadIdx.x;
  if (t >= E_HH) return;
  atomicAdd(&cnt[dst[t]], 1.0f);
}
__global__ void gat_edge_logits(const int* __restrict__ src, const int* __restrict__ dst,
                                const float* __restrict__ xl, const float* __restrict__ xr,
                                const float* __restrict__ att, float* __restrict__ elog,
                                float* __restrict__ m, int outc, int heads) {
  int t = blockIdx.x * blockDim.x + threadIdx.x;
  if (t >= E_HH + N_HIGH) return;
  int s, d;
  if (t < E_HH) { s = src[t]; d = dst[t]; } else { s = d = t - E_HH; }
  for (int h = 0; h < heads; ++h) {
    const float4* xls = (const float4*)(xl + (size_t)s * outc + h * 16);
    const float4* xrd = (const float4*)(xr + (size_t)d * outc + h * 16);
    const float4* at4 = (const float4*)(att + h * 16);
    float acc = 0.0f;
#pragma unroll
    for (int q = 0; q < 4; ++q) {
      float4 a = xls[q], b = xrd[q], w = at4[q];
      float v0 = a.x + b.x, v1 = a.y + b.y, v2 = a.z + b.z, v3 = a.w + b.w;
      v0 = (v0 > 0.0f) ? v0 : 0.2f * v0; v1 = (v1 > 0.0f) ? v1 : 0.2f * v1;
      v2 = (v2 > 0.0f) ? v2 : 0.2f * v2; v3 = (v3 > 0.0f) ? v3 : 0.2f * v3;
      acc += w.x*v0 + w.y*v1 + w.z*v2 + w.w*v3;
    }
    elog[(size_t)t * heads + h] = acc;
    atomicMaxFloat(&m[d * heads + h], acc);
  }
}
__global__ void gat_edge_exp(const int* __restrict__ dst, float* __restrict__ elog,
                             const float* __restrict__ m, float* __restrict__ den, int heads) {
  int t = blockIdx.x * blockDim.x + threadIdx.x;
  if (t >= E_HH + N_HIGH) return;
  int d = (t < E_HH) ? dst[t] : (t - E_HH);
  for (int h = 0; h < heads; ++h) {
    float a = __expf(elog[(size_t)t * heads + h] - m[d * heads + h]);
    elog[(size_t)t * heads + h] = a;
    atomicAdd(&den[d * heads + h], a);
  }
}
__global__ void gat_edge_agg(const int* __restrict__ src, const int* __restrict__ dst,
                             const float* __restrict__ xl, const float* __restrict__ elog,
                             const float* __restrict__ den, float* __restrict__ agg,
                             int outc, int heads) {
  int t = blockIdx.x * blockDim.x + threadIdx.x;
  if (t >= E_HH + N_HIGH) return;
  int s, d;
  if (t < E_HH) { s = src[t]; d = dst[t]; } else { s = d = t - E_HH; }
  for (int h = 0; h < heads; ++h) {
    float alpha = elog[(size_t)t * heads + h] / den[d * heads + h];
    const float4* xls = (const float4*)(xl + (size_t)s * outc + h * 16);
    float* ad = agg + (size_t)d * outc + h * 16;
#pragma unroll
    for (int q = 0; q < 4; ++q) {
      float4 a = xls[q];
      atomicAdd(&ad[4*q+0], alpha * a.x);
      atomicAdd(&ad[4*q+1], alpha * a.y);
      atomicAdd(&ad[4*q+2], alpha * a.z);
      atomicAdd(&ad[4*q+3], alpha * a.w);
    }
  }
}
__global__ void gat_finalize(const float* __restrict__ agg, const float* __restrict__ cnt,
                             const float* __restrict__ bias, float* __restrict__ h,
                             int outc, int relu) {
  int t = blockIdx.x * blockDim.x + threadIdx.x;
  if (t >= N_HIGH * 32) return;
  int c = t & 31, n = t >> 5;
  float v = 0.0f;
  if (c < outc) {
    v = agg[(size_t)n * outc + c] / cnt[n] + bias[c];
    if (relu) v = fmaxf(v, 0.0f);
  }
  h[t] = v;
}
__global__ void pred_out(const float* __restrict__ t2, const float* __restrict__ W3,
                         const float* __restrict__ b3, float* __restrict__ out) {
  int i = blockIdx.x * blockDim.x + threadIdx.x;
  if (i >= N_HIGH) return;
  float4 a = ((const float4*)(t2 + (size_t)i * 16))[0];
  float4 b = ((const float4*)(t2 + (size_t)i * 16))[1];
  float4 w0 = ((const float4*)W3)[0];
  float4 w1 = ((const float4*)W3)[1];
  out[i] = b3[0] + a.x*w0.x + a.y*w0.y + a.z*w0.z + a.w*w0.w
                 + b.x*w1.x + b.y*w1.y + b.z*w1.z + b.w*w1.w;
}

// ---------------- host launcher ----------------
extern "C" void kernel_launch(void* const* d_in, const int* in_sizes, int n_in,
                              void* d_out, int out_size, void* d_ws, size_t ws_size,
                              hipStream_t stream) {
  (void)in_sizes; (void)n_in; (void)out_size; (void)ws_size;
  const float* x_low  = (const float*)d_in[0];
  const float* x_high = (const float*)d_in[1];
  const float* z_std  = (const float*)d_in[2];
  const int*   e_ll   = (const int*)d_in[3];
  const int*   e_lh   = (const int*)d_in[4];
  const int*   e_hh   = (const int*)d_in[5];
  const float* att_p  = (const float*)d_in[6];
  const float* Wz  = (const float*)d_in[7];   const float* bz  = (const float*)d_in[8];
  const float* Wh  = (const float*)d_in[9];   const float* bhh = (const float*)d_in[10];
  const float* Wlz = (const float*)d_in[11];  const float* blz = (const float*)d_in[12];
  const float* Wlh = (const float*)d_in[13];  const float* blh = (const float*)d_in[14];
  const float* Wd  = (const float*)d_in[15];  const float* bd  = (const float*)d_in[16];
  const float* Wdr = (const float*)d_in[17];  const float* bdr = (const float*)d_in[18];
  const float* datt = (const float*)d_in[19]; const float* dbias = (const float*)d_in[20];
  const float* bn0g = (const float*)d_in[21]; const float* bn0b = (const float*)d_in[22];
  static const int WLI[5] = {23,31,39,47,55}, BLI[5] = {24,32,40,48,56};
  static const int WRI[5] = {25,33,41,49,57}, BRI[5] = {26,34,42,50,58};
  static const int ATTI[5] = {27,35,43,51,59}, BII[5] = {28,36,44,52,60};
  static const int BNG[4] = {29,37,45,53}, BNB[4] = {30,38,46,54};
  const float* pW1  = (const float*)d_in[61]; const float* pb1  = (const float*)d_in[62];
  const float* pbn1g = (const float*)d_in[63]; const float* pbn1b = (const float*)d_in[64];
  const float* pW2  = (const float*)d_in[65]; const float* pb2  = (const float*)d_in[66];
  const float* pbn2g = (const float*)d_in[67]; const float* pbn2b = (const float*)d_in[68];
  const float* pW3  = (const float*)d_in[69]; const float* pb3  = (const float*)d_in[70];

  // workspace bump allocator
  char* ws = (char*)d_ws;
  size_t off = 0;
  auto carve = [&](size_t bytes) -> char* {
    off = (off + 255) & ~(size_t)255;
    char* p = ws + off; off += bytes; return p;
  };
  float* probs = (float*)carve(32 * 4);
  float* dinv  = (float*)carve((size_t)N_LOW * 4);
  float* enc   = (float*)carve((size_t)N_LOW * 32 * 4);
  unsigned short* encb = (unsigned short*)carve((size_t)N_LOW * 32 * 2);
  float* xllow = (float*)carve((size_t)N_LOW * 64 * 4);
  unsigned short* wz_b  = (unsigned short*)carve(32 * 32 * 2);
  unsigned short* wh_b  = (unsigned short*)carve(32 * 32 * 2);
  unsigned short* wlz_b = (unsigned short*)carve(32 * 32 * 2);
  unsigned short* wlh_b = (unsigned short*)carve(32 * 32 * 2);
  unsigned short* wd_b  = (unsigned short*)carve(32 * 64 * 2);
  unsigned short* wl_b  = (unsigned short*)carve(32 * 32 * 2);
  unsigned short* wr_b  = (unsigned short*)carve(32 * 32 * 2);
  float* bnsum = (float*)carve(32 * 4);
  float* bnsq  = (float*)carve(32 * 4);
  float* cnt   = (float*)carve((size_t)N_HIGH * 4);
  float* mbuf  = (float*)carve((size_t)N_HIGH * 2 * 4);
  float* den   = (float*)carve((size_t)N_HIGH * 2 * 4);
  char* bigA = carve((size_t)32 << 20);  // xmsg f32 -> hz f32 -> [h | xl]
  char* bigB = carve((size_t)16 << 20);  // xmsgb -> hb
  char* bigC = carve((size_t)32 << 20);  // hh f32 -> [xr | agg]
  char* bigD = carve((size_t)16 << 20);  // hzb -> elog -> y1
  char* bigE = carve((size_t)16 << 20);  // hhb -> t2
  float* xmsg = (float*)bigA;
  unsigned short* xmsgb = (unsigned short*)bigB;
  float* hz  = (float*)bigA;
  float* hhf = (float*)bigC;
  unsigned short* hzb = (unsigned short*)bigD;
  unsigned short* hhb = (unsigned short*)bigE;
  float* h   = (float*)bigA;
  float* xl  = (float*)(bigA + ((size_t)14 << 20));
  unsigned short* hb = (unsigned short*)bigB;
  float* xr  = (float*)bigC;
  float* agg = (float*)(bigC + ((size_t)14 << 20));
  float* elog = (float*)bigD;
  float* y1   = (float*)bigD;
  float* t2   = (float*)bigE;

  const int B = 256;
  const int MROWS = N_LOW * PERIODS;      // 250000
  const int MtA = MROWS / 16;             // 15625
  const int MtH = N_HIGH / 16;            // 6250

  // ---- Stage A: A3TGCN ----
  fill_f32<<<DIVUP(N_LOW, B), B, 0, stream>>>(dinv, 0.0f, N_LOW);
  deg_count<<<DIVUP(E_LL + N_LOW, B), B, 0, stream>>>(e_ll, dinv);
  deg_to_dinv<<<DIVUP(N_LOW, B), B, 0, stream>>>(dinv);
  softmax_periods<<<1, 32, 0, stream>>>(att_p, probs);
  xmsg_init<<<DIVUP(MROWS * 32, B), B, 0, stream>>>(x_low, dinv, xmsg);
  xmsg_scatter<<<DIVUP(E_LL * PERIODS, B), B, 0, stream>>>(x_low, e_ll, dinv, xmsg);
  f32_to_bf16_vec4<<<DIVUP(MROWS * 8, B), B, 0, stream>>>(xmsgb, xmsg, MROWS * 8);
  pack_bfrag<<<DIVUP(32 * 32, B), B, 0, stream>>>(wz_b, Wz, LOW_IN, LOW_IN, 32);
  pack_bfrag<<<DIVUP(32 * 32, B), B, 0, stream>>>(wh_b, Wh, LOW_IN, LOW_IN, 32);
  gemm_bf16<<<DIVUP(MtA * 2 * 32, B), B, 0, stream>>>(xmsgb, wz_b, bz, LOW_IN, hz, 32, MtA, 32, 0);
  gemm_bf16<<<DIVUP(MtA * 2 * 32, B), B, 0, stream>>>(xmsgb, wh_b, bhh, LOW_IN, hhf, 32, MtA, 32, 0);
  f32_to_bf16_vec4<<<DIVUP(MROWS * 8, B), B, 0, stream>>>(hzb, hz, MROWS * 8);
  f32_to_bf16_vec4<<<DIVUP(MROWS * 8, B), B, 0, stream>>>(hhb, hhf, MROWS * 8);
  pack_bfrag<<<DIVUP(32 * 32, B), B, 0, stream>>>(wlz_b, Wlz, LOW_IN, LOW_IN, 32);
  pack_bfrag<<<DIVUP(32 * 32, B), B, 0, stream>>>(wlh_b, Wlh, LOW_IN, LOW_IN, 32);
  fill_f32<<<DIVUP(N_LOW * 32, B), B, 0, stream>>>(enc, 0.0f, N_LOW * 32);
  tgcn_gate_accum<<<DIVUP(MtA * 32, B), B, 0, stream>>>(hzb, hhb, wlz_b, wlh_b, blz, blh, probs, enc, MtA);

  // ---- Stage B: downscaler ----
  f32_to_bf16_vec4<<<DIVUP(N_LOW * 8, B), B, 0, stream>>>(encb, enc, N_LOW * 8);
  pack_bfrag<<<DIVUP(32 * 64, B), B, 0, stream>>>(wd_b, Wd, LOW_IN, 64, 64);
  gemm_bf16<<<DIVUP((N_LOW / 16) * 4 * 32, B), B, 0, stream>>>(encb, wd_b, bd, 64, xllow, 64, N_LOW / 16, 64, 0);
  h0_init<<<DIVUP(N_HIGH * 32, B), B, 0, stream>>>(z_std, h);
  downscale_gat<<<DIVUP(N_HIGH, B), B, 0, stream>>>(xllow, e_lh, x_high, Wdr, bdr, datt, dbias, h);
  fill_f32<<<1, 64, 0, stream>>>(bnsum, 0.0f, 32);
  fill_f32<<<1, 64, 0, stream>>>(bnsq, 0.0f, 32);
  bn_reduce<<<64, B, 0, stream>>>(h, N_HIGH, 17, 32, bnsum, bnsq);
  bn_norm<<<DIVUP(N_HIGH * 17, B), B, 0, stream>>>(h, N_HIGH, 17, 32, bnsum, bnsq, bn0g, bn0b, 0);

  // in-degree (+self loop) counts, constant across layers
  fill_f32<<<DIVUP(N_HIGH, B), B, 0, stream>>>(cnt, 1.0f, N_HIGH);
  cnt_count<<<DIVUP(E_HH, B), B, 0, stream>>>(e_hh + E_HH, cnt);

  // ---- high-graph GATv2 layers ----
  const int NT = E_HH + N_HIGH;
  for (int L = 0; L < 5; ++L) {
    int heads = (L == 4) ? 1 : 2;
    int oc = heads * 16;
    int ic = (L == 0) ? 17 : 32;
    const float* pWl = (const float*)d_in[WLI[L]];
    const float* pbl = (const float*)d_in[BLI[L]];
    const float* pWr = (const float*)d_in[WRI[L]];
    const float* pbr = (const float*)d_in[BRI[L]];
    const float* patt = (const float*)d_in[ATTI[L]];
    const float* pbias = (const float*)d_in[BII[L]];
    f32_to_bf16_vec4<<<DIVUP(N_HIGH * 8, B), B, 0, stream>>>(hb, h, N_HIGH * 8);
    pack_bfrag<<<DIVUP(32 * oc, B), B, 0, stream>>>(wl_b, pWl, ic, oc, oc);
    gemm_bf16<<<DIVUP(MtH * (oc / 16) * 32, B), B, 0, stream>>>(hb, wl_b, pbl, oc, xl, oc, MtH, oc, 0);
    pack_bfrag<<<DIVUP(32 * oc, B), B, 0, stream>>>(wr_b, pWr, ic, oc, oc);
    gemm_bf16<<<DIVUP(MtH * (oc / 16) * 32, B), B, 0, stream>>>(hb, wr_b, pbr, oc, xr, oc, MtH, oc, 0);
    fill_f32<<<DIVUP(N_HIGH * heads, B), B, 0, stream>>>(mbuf, -3.0e38f, N_HIGH * heads);
    fill_f32<<<DIVUP(N_HIGH * heads, B), B, 0, stream>>>(den, 0.0f, N_HIGH * heads);
    fill_f32<<<DIVUP(N_HIGH * oc, B), B, 0, stream>>>(agg, 0.0f, N_HIGH * oc);
    gat_edge_logits<<<DIVUP(NT, B), B, 0, stream>>>(e_hh, e_hh + E_HH, xl, xr, patt, elog, mbuf, oc, heads);
    gat_edge_exp<<<DIVUP(NT, B), B, 0, stream>>>(e_hh + E_HH, elog, mbuf, den, heads);
    gat_edge_agg<<<DIVUP(NT, B), B, 0, stream>>>(e_hh, e_hh + E_HH, xl, elog, den, agg, oc, heads);
    gat_finalize<<<DIVUP(N_HIGH * 32, B), B, 0, stream>>>(agg, cnt, pbias, h, oc, (L == 4) ? 1 : 0);
    if (L < 4) {
      fill_f32<<<1, 64, 0, stream>>>(bnsum, 0.0f, 32);
      fill_f32<<<1, 64, 0, stream>>>(bnsq, 0.0f, 32);
      bn_reduce<<<64, B, 0, stream>>>(h, N_HIGH, 32, 32, bnsum, bnsq);
      bn_norm<<<DIVUP(N_HIGH * 32, B), B, 0, stream>>>(h, N_HIGH, 32, 32, bnsum, bnsq,
                                                       (const float*)d_in[BNG[L]],
                                                       (const float*)d_in[BNB[L]], 1);
    }
  }

  // ---- prediction head ----
  f32_to_bf16_vec4<<<DIVUP(N_HIGH * 8, B), B, 0, stream>>>(hb, h, N_HIGH * 8);
  pack_bfrag<<<DIVUP(32 * 16, B), B, 0, stream>>>(wl_b, pW1, 16, 16, 16);
  gemm_bf16<<<DIVUP(MtH * 1 * 32, B), B, 0, stream>>>(hb, wl_b, pb1, 16, y1, 16, MtH, 16, 1);
  fill_f32<<<1, 64, 0, stream>>>(bnsum, 0.0f, 32);
  fill_f32<<<1, 64, 0, stream>>>(bnsq, 0.0f, 32);
  bn_reduce<<<64, B, 0, stream>>>(y1, N_HIGH, 16, 16, bnsum, bnsq);
  bn_norm<<<DIVUP(N_HIGH * 16, B), B, 0, stream>>>(y1, N_HIGH, 16, 16, bnsum, bnsq, pbn1g, pbn1b, 0);
  f32_to_bf16_pad<<<DIVUP(N_HIGH * 32, B), B, 0, stream>>>(hb, y1, N_HIGH, 32, N_HIGH, 16);
  pack_bfrag<<<DIVUP(32 * 16, B), B, 0, stream>>>(wr_b, pW2, 16, 8, 16);
  gemm_bf16<<<DIVUP(MtH * 1 * 32, B), B, 0, stream>>>(hb, wr_b, pb2, 8, t2, 16, MtH, 16, 1);
  fill_f32<<<1, 64, 0, stream>>>(bnsum, 0.0f, 32);
  fill_f32<<<1, 64, 0, stream>>>(bnsq, 0.0f, 32);
  bn_reduce<<<64, B, 0, stream>>>(t2, N_HIGH, 8, 16, bnsum, bnsq);
  bn_norm<<<DIVUP(N_HIGH * 8, B), B, 0, stream>>>(t2, N_HIGH, 8, 16, bnsum, bnsq, pbn2g, pbn2b, 0);
  pred_out<<<DIVUP(N_HIGH, B), B, 0, stream>>>(t2, pW3, pb3, (float*)d_out);
}
